// RVQQuantizer_65472481460430
// MI455X (gfx1250) — compile-verified, matrix operand order checked by reading, not run on hardware
//
#include <hip/hip_runtime.h>
#include <math.h>

typedef __attribute__((ext_vector_type(16))) __bf16 v16bf;
typedef __attribute__((ext_vector_type(8)))  float  v8f;
typedef __attribute__((ext_vector_type(4)))  int    v4i;

#define BT      16384            // B*T = 8*2048 rows
#define DD      256              // feature dim
#define KCB     4096             // codebook entries
#define NTILES  (KCB / 16)       // 256 column tiles
#define KSTEPS  (DD / 32)        // 8 k-steps of 32 for bf16 WMMA
#define FRAG_ELEMS (KSTEPS * 32 * 16)  // 4096 bf16 elems per 16-col (or 16-row) tile
#define EPSV    1e-10f

#define CHUNK_NT  2                          // column tiles per LDS chunk
#define NCHUNKS   (NTILES / CHUNK_NT)        // 128
#define CHUNK_BF  (CHUNK_NT * FRAG_ELEMS)    // 8192 bf16 per precision (16 KB)

// ---- workspace layout (bytes) ----
#define SZ_ZFRAG   ((size_t)BT * DD * 2)     // 8 MB per precision
#define SZ_CBFRAG  ((size_t)KCB * DD * 2)    // 2 MB per precision
#define OFF_ZHI    ((size_t)0)
#define OFF_ZLO    (OFF_ZHI + SZ_ZFRAG)
#define OFF_CBHI   (OFF_ZLO + SZ_ZFRAG)
#define OFF_CBLO   (OFF_CBHI + SZ_CBFRAG)
#define OFF_ESQ    (OFF_CBLO + SZ_CBFRAG)
#define OFF_IDX    (OFF_ESQ + (size_t)KCB * 4)
#define OFF_LOSS   (OFF_IDX + (size_t)BT * 4)

// ---- output layout (float elements) ----
#define OUT_ZQ    0
#define OUT_IDX   ((size_t)BT * DD)          // 4194304
#define OUT_LOSS  (OUT_IDX + BT)             // 4210688
#define OUT_PERP  (OUT_LOSS + 1)
#define OUT_ENT   (OUT_PERP + 1)

// ---- CDNA5 async memory->LDS path (no VGPR round-trip, ASYNCcnt tracked) ----
#if defined(__has_builtin)
#if __has_builtin(__builtin_amdgcn_global_load_async_to_lds_b128)
#define HAVE_ASYNC_LDS 1
#endif
#endif

__device__ __forceinline__ void stage_b128(const void* g, void* l) {
#ifdef HAVE_ASYNC_LDS
  __builtin_amdgcn_global_load_async_to_lds_b128(
      (__attribute__((address_space(1))) v4i*)(g),
      (__attribute__((address_space(3))) v4i*)(l), 0, 0);
#else
  *(uint4*)l = *(const uint4*)g;
#endif
}

__device__ __forceinline__ void wait_async_le8() {
#ifdef HAVE_ASYNC_LDS
#if __has_builtin(__builtin_amdgcn_s_wait_asynccnt)
  __builtin_amdgcn_s_wait_asynccnt(8);
#else
  asm volatile("s_wait_asynccnt 8" ::: "memory");
#endif
#endif
}

__device__ __forceinline__ void wait_async_zero() {
#ifdef HAVE_ASYNC_LDS
#if __has_builtin(__builtin_amdgcn_s_wait_asynccnt)
  __builtin_amdgcn_s_wait_asynccnt(0);
#else
  asm volatile("s_wait_asynccnt 0" ::: "memory");
#endif
#endif
}

// Fragment address for matrix element (row tile `tile`, in-tile row r15, K index kk+32*kstep):
//   lane = r15 + (kk&8 ? 16 : 0); elem = (kk&7) + (kk&16 ? 8 : 0)
//   addr = ((tile*KSTEPS + kstep)*32 + lane)*16 + elem
// Matches CDNA5 16-bit A/B VGPR striping so the GEMM kernel loads v16bf directly.

// ---------- prep: codebook -> bf16 hi/lo fragments + row norms ----------
__global__ void vq_prep_codebook(const float* __restrict__ cb,
                                 __bf16* __restrict__ cbhi, __bf16* __restrict__ cblo,
                                 float* __restrict__ esq) {
  const int tid  = threadIdx.x;
  const int lane = tid & 31;
  const int wave = tid >> 5;
  const int r     = blockIdx.x * 8 + wave;          // codebook row
  const int tile  = r >> 4;
  const int flane = (r & 15) + ((lane & 8) ? 16 : 0);
  const int e     = (lane & 7) + ((lane & 16) ? 8 : 0);
  float sq = 0.f;
#pragma unroll
  for (int j = 0; j < KSTEPS; ++j) {               // K = lane + 32*j  -> kk = lane, kstep = j
    float x = cb[(size_t)r * DD + lane + j * 32];
    sq += x * x;
    __bf16 h = (__bf16)x;
    __bf16 l = (__bf16)(x - (float)h);
    const size_t a = ((size_t)(tile * KSTEPS + j) * 32 + flane) * 16 + e;
    cbhi[a] = h;
    cblo[a] = l;
  }
#pragma unroll
  for (int off = 16; off >= 1; off >>= 1) sq += __shfl_xor(sq, off, 32);
  if (lane == 0) esq[r] = sq;
}

// ---------- prep: z -> bf16 hi/lo fragments ----------
__global__ void vq_prep_z(const float* __restrict__ z,
                          __bf16* __restrict__ zhi, __bf16* __restrict__ zlo) {
  const int tid  = threadIdx.x;
  const int lane = tid & 31;
  const int wave = tid >> 5;
  const int r     = blockIdx.x * 8 + wave;          // z row (0..BT-1)
  const int tile  = r >> 4;
  const int flane = (r & 15) + ((lane & 8) ? 16 : 0);
  const int e     = (lane & 7) + ((lane & 16) ? 8 : 0);
#pragma unroll
  for (int j = 0; j < KSTEPS; ++j) {
    float x = z[(size_t)r * DD + lane + j * 32];
    __bf16 h = (__bf16)x;
    __bf16 l = (__bf16)(x - (float)h);
    const size_t a = ((size_t)(tile * KSTEPS + j) * 32 + flane) * 16 + e;
    zhi[a] = h;
    zlo[a] = l;
  }
}

// ---------- main: WMMA distance GEMM + running argmin, double-buffered async LDS ----------
__global__ void __launch_bounds__(256) vq_argmin_wmma(
    const __bf16* __restrict__ zhi, const __bf16* __restrict__ zlo,
    const __bf16* __restrict__ cbhi, const __bf16* __restrict__ cblo,
    const float* __restrict__ esq,
    int* __restrict__ idx_out, float* __restrict__ idx_f_out) {
  // 2 buffers x (hi 16KB | lo 16KB) = 64 KB (WGP has 320 KB LDS)
  __shared__ __align__(16) __bf16 lds[2][2 * CHUNK_BF];
  const int tid   = threadIdx.x;
  const int lane  = tid & 31;
  const int wave  = tid >> 5;
  const int mtile = blockIdx.x * 8 + wave;                // 16-row tile owned by this wave

  // A panel (16 rows x 256 K, hi+lo) resident in registers for the whole kernel
  const v16bf* zh = (const v16bf*)zhi + (size_t)mtile * KSTEPS * 32 + lane;
  const v16bf* zl = (const v16bf*)zlo + (size_t)mtile * KSTEPS * 32 + lane;
  v16bf ah[KSTEPS], al[KSTEPS];
#pragma unroll
  for (int k = 0; k < KSTEPS; ++k) { ah[k] = zh[k * 32]; al[k] = zl[k * 32]; }

  float bestD[8];
  int   bestI[8];
#pragma unroll
  for (int v = 0; v < 8; ++v) { bestD[v] = 3.4e38f; bestI[v] = 0; }

  // stage helper: copy chunk c (hi+lo, 32 KB) into buffer `buf`; 8 async b128 per thread
  auto stage = [&](int buf, int c) {
    const char* srcH = (const char*)(cbhi + (size_t)c * CHUNK_BF);
    const char* srcL = (const char*)(cblo + (size_t)c * CHUNK_BF);
    char* dstH = (char*)&lds[buf][0];
    char* dstL = (char*)&lds[buf][CHUNK_BF];
#pragma unroll
    for (int i = 0; i < 4; ++i) {
      const int off = (tid + i * 256) * 16;
      stage_b128(srcH + off, dstH + off);
      stage_b128(srcL + off, dstL + off);
    }
  };

  stage(0, 0);
  for (int c = 0; c < NCHUNKS; ++c) {
    const int buf = c & 1;
    if (c + 1 < NCHUNKS) {
      stage(buf ^ 1, c + 1);   // prefetch next chunk behind the WMMAs
      wait_async_le8();        // in-order completion: first 8 (chunk c) are done
    } else {
      wait_async_zero();
    }
    __syncthreads();           // all waves' async LDS writes for chunk c visible

#pragma unroll
    for (int ct = 0; ct < CHUNK_NT; ++ct) {
      const int ntile = c * CHUNK_NT + ct;
      const v16bf* Bh = (const v16bf*)&lds[buf][0]        + ct * KSTEPS * 32 + lane;
      const v16bf* Bl = (const v16bf*)&lds[buf][CHUNK_BF] + ct * KSTEPS * 32 + lane;
      v8f acc = {};
#pragma unroll
      for (int k = 0; k < KSTEPS; ++k) {
        v16bf bh = Bh[k * 32];
        v16bf bl = Bl[k * 32];
        // 3-term bf16 split of the f32 dot product, f32 accumulation
        acc = __builtin_amdgcn_wmma_f32_16x16x32_bf16(false, ah[k], false, bh, (short)0, acc, false, false);
        acc = __builtin_amdgcn_wmma_f32_16x16x32_bf16(false, al[k], false, bh, (short)0, acc, false, false);
        acc = __builtin_amdgcn_wmma_f32_16x16x32_bf16(false, ah[k], false, bl, (short)0, acc, false, false);
      }
      const int   n  = ntile * 16 + (lane & 15);
      const float es = esq[n];
#pragma unroll
      for (int v = 0; v < 8; ++v) {
        float d = es - 2.0f * acc[v];   // ||z||^2 omitted: constant per row
        if (d < bestD[v]) { bestD[v] = d; bestI[v] = n; }
      }
    }
    __syncthreads();           // done reading buf before it is restaged
  }

  // cross-lane argmin within each 16-lane group (rows v and v+8), first-min tie-break
#pragma unroll
  for (int v = 0; v < 8; ++v) {
    float d = bestD[v];
    int   i = bestI[v];
#pragma unroll
    for (int off = 8; off >= 1; off >>= 1) {
      float od = __shfl_xor(d, off, 16);
      int   oi = __shfl_xor(i, off, 16);
      if (od < d || (od == d && oi < i)) { d = od; i = oi; }
    }
    if ((lane & 15) == 0) {
      int m = mtile * 16 + v + ((lane >= 16) ? 8 : 0);
      idx_out[m]   = i;
      idx_f_out[m] = (float)i;
    }
  }
}

// ---------- gather z_q = codebook[idx] + commitment-loss partial ----------
__global__ void vq_gather_loss(const float* __restrict__ z, const float* __restrict__ cb,
                               const int* __restrict__ idx, float* __restrict__ zq,
                               float* __restrict__ loss_acc) {
  const int tid  = threadIdx.x;
  const int lane = tid & 31;
  const int wave = tid >> 5;
  const int m = blockIdx.x * 8 + wave;
  const int r = idx[m];
  const float* crow = cb + (size_t)r * DD;
  const float* zrow = z  + (size_t)m * DD;
  float* qrow = zq + (size_t)m * DD;
  float s = 0.f;
#pragma unroll
  for (int j = 0; j < 8; ++j) {
    const int c = lane + j * 32;
    float e  = crow[c];
    float zz = zrow[c];
    qrow[c] = e;                 // straight-through forward value == z_q
    float df = e - zz;
    s += df * df;
  }
#pragma unroll
  for (int off = 16; off >= 1; off >>= 1) s += __shfl_xor(s, off, 32);
  if (lane == 0) atomicAdd(loss_acc, s);
}

// ---------- usage entropy / perplexity (single block) ----------
__global__ void vq_usage_stats(const float* __restrict__ usage,
                               float* __restrict__ out_perp, float* __restrict__ out_ent) {
  __shared__ float red[256];
  const int tid = threadIdx.x;
  float s = 0.f;
  for (int i = tid; i < KCB; i += 256) s += usage[i];
  red[tid] = s; __syncthreads();
  for (int w = 128; w >= 1; w >>= 1) { if (tid < w) red[tid] += red[tid + w]; __syncthreads(); }
  const float total = red[0];
  __syncthreads();
  float ep = 0.f;
  for (int i = tid; i < KCB; i += 256) {
    float p = usage[i] / (total + EPSV);
    ep += -p * logf(p + EPSV);
  }
  red[tid] = ep; __syncthreads();
  for (int w = 128; w >= 1; w >>= 1) { if (tid < w) red[tid] += red[tid + w]; __syncthreads(); }
  if (tid == 0) {
    float ent = red[0];
    *out_ent  = ent;
    *out_perp = expf(ent);
  }
}

// ---------- finalize mean loss ----------
__global__ void vq_finalize_loss(const float* __restrict__ loss_acc, float* __restrict__ out_loss) {
  *out_loss = *loss_acc / (float)((size_t)BT * DD);
}

extern "C" void kernel_launch(void* const* d_in, const int* in_sizes, int n_in,
                              void* d_out, int out_size, void* d_ws, size_t ws_size,
                              hipStream_t stream) {
  (void)in_sizes; (void)n_in; (void)out_size; (void)ws_size;
  const float* z     = (const float*)d_in[0];   // [8,2048,256]
  const float* cb    = (const float*)d_in[1];   // [4096,256]
  const float* usage = (const float*)d_in[2];   // [4096]
  float* out = (float*)d_out;
  char*  ws  = (char*)d_ws;

  __bf16* zhi  = (__bf16*)(ws + OFF_ZHI);
  __bf16* zlo  = (__bf16*)(ws + OFF_ZLO);
  __bf16* cbhi = (__bf16*)(ws + OFF_CBHI);
  __bf16* cblo = (__bf16*)(ws + OFF_CBLO);
  float*  esq  = (float*)(ws + OFF_ESQ);
  int*    idxw = (int*)(ws + OFF_IDX);
  float*  lacc = (float*)(ws + OFF_LOSS);

  (void)hipMemsetAsync(lacc, 0, sizeof(float), stream);

  vq_prep_codebook<<<KCB / 8, 256, 0, stream>>>(cb, cbhi, cblo, esq);
  vq_prep_z       <<<BT  / 8, 256, 0, stream>>>(z, zhi, zlo);
  vq_argmin_wmma  <<<BT / 128, 256, 0, stream>>>(zhi, zlo, cbhi, cblo, esq,
                                                 idxw, out + OUT_IDX);
  vq_gather_loss  <<<BT / 8, 256, 0, stream>>>(z, cb, idxw, out + OUT_ZQ, lacc);
  vq_usage_stats  <<<1, 256, 0, stream>>>(usage, out + OUT_PERP, out + OUT_ENT);
  vq_finalize_loss<<<1, 1, 0, stream>>>(lacc, out + OUT_LOSS);
}
